// CustomTransformer2D_88888643158586
// MI455X (gfx1250) — matrix operand order repaired and checked by reading, not compiled, not run on hardware
//
#include <hip/hip_runtime.h>

typedef __attribute__((ext_vector_type(16))) _Float16 v16h;
typedef __attribute__((ext_vector_type(8)))  _Float16 v8h;
typedef __attribute__((ext_vector_type(8)))  float    v8f;

#define TILE_T 8
#define ROWS   128           // TILE_T * 16 neighbors
#define EPS    1e-5f

// ---------- WMMA fragment helpers (layouts per CDNA5 ISA 7.12.2) ----------

__device__ __forceinline__ v16h make_frag(const _Float16* p0, const _Float16* p1) {
  v8h a = *(const v8h*)p0;
  v8h b = *(const v8h*)p1;
  v16h r;
#pragma unroll
  for (int i = 0; i < 8; ++i) { r[i] = a[i]; r[i + 8] = b[i]; }
  return r;
}

// A: 16x32 f16 tile at (rowBase, kOff) of a row-major [.. x stride] f16 array.
// lane m = lane&15; halves 0..7 = A[m][kOff + hi*8 .. +7], halves 8..15 = +16.
__device__ __forceinline__ v16h load_a_frag(const _Float16* base, int stride,
                                            int rowBase, int kOff, int l15, int hi) {
  const _Float16* p = base + (size_t)(rowBase + l15) * stride + kOff + hi * 8;
  return make_frag(p, p + 16);
}

// B: 32x16 built from row-major weight W[out][in]; B[k][n] = W[outBase+n][k].
// lane col n = lane&15; halves 0..7 = W row at kOff + hi*8 .. +7, halves 8..15 = +16.
__device__ __forceinline__ v16h load_b_frag(const _Float16* w, int din,
                                            int outBase, int kOff, int l15, int hi) {
  const _Float16* p = w + (size_t)(outBase + l15) * din + kOff + hi * 8;
  return make_frag(p, p + 16);
}

__device__ __forceinline__ v8f wmma16(v16h a, v16h b, v8f c) {
  return __builtin_amdgcn_wmma_f32_16x16x32_f16(false, a, false, b, (short)0, c,
                                                false, false);
}

// ---------- weight fp32 -> fp16 conversion into d_ws ----------
// layout (halves): Wq 4096 | Wk 4096 | Wv 4096 | Wo 4096 | W1 16384 | W2 16384
__global__ __launch_bounds__(256) void convert_weights(
    const float* __restrict__ Wq, const float* __restrict__ Wk,
    const float* __restrict__ Wv, const float* __restrict__ Wo,
    const float* __restrict__ W1, const float* __restrict__ W2,
    _Float16* __restrict__ out) {
  int i = blockIdx.x * 256 + threadIdx.x;   // 0..49151
  float v;
  if      (i <  4096) v = Wq[i];
  else if (i <  8192) v = Wk[i - 4096];
  else if (i < 12288) v = Wv[i - 8192];
  else if (i < 16384) v = Wo[i - 12288];
  else if (i < 32768) v = W1[i - 16384];
  else                v = W2[i - 32768];
  out[i] = (_Float16)v;
}

// ---------- fully fused transformer block ----------
__global__ __launch_bounds__(256) void te_fused(
    const float* __restrict__ qg,  const float* __restrict__ kg,
    const float* __restrict__ posg,
    const float* __restrict__ bq,  const float* __restrict__ bk,
    const float* __restrict__ bv,
    const float* __restrict__ Wpg, const float* __restrict__ bpg,
    const float* __restrict__ bo,  const float* __restrict__ b1,
    const float* __restrict__ b2,
    const float* __restrict__ ln1g, const float* __restrict__ ln1b,
    const float* __restrict__ ln2g, const float* __restrict__ ln2b,
    const _Float16* __restrict__ wgt, float* __restrict__ outg) {

  __shared__ __align__(16) _Float16 sA [ROWS * 64];   // k16, then v16
  __shared__ __align__(16) _Float16 sKP[ROWS * 64];   // kp16, then h1 [16][256]
  __shared__ __align__(16) float    sQ  [TILE_T * 64];
  __shared__ __align__(16) _Float16 sQN [16 * 64];    // padded to 16 rows
  __shared__ __align__(16) float    sQP [TILE_T * 64];
  __shared__ __align__(16) float    sPOS[ROWS * 4];
  __shared__ __align__(16) float    sWp [64 * 4];
  __shared__             float      sBp [64];
  __shared__ __align__(16) _Float16 sO  [16 * 64];    // padded
  __shared__ __align__(16) float    sX  [TILE_T * 64];
  __shared__ __align__(16) _Float16 sXN [16 * 64];    // padded

  const int tid  = threadIdx.x;
  const int wid  = tid >> 5;
  const int lane = tid & 31;
  const int l15  = lane & 15;
  const int hi   = lane >> 4;
  const size_t tokenBase = (size_t)blockIdx.x * TILE_T;

  const _Float16* wWq = wgt;
  const _Float16* wWk = wgt + 4096;
  const _Float16* wWv = wgt + 8192;
  const _Float16* wWo = wgt + 12288;
  const _Float16* wW1 = wgt + 16384;
  const _Float16* wW2 = wgt + 32768;

  // ---- stage tile inputs ----
  {
    if (tid < 128) {
      ((float4*)sQ)[tid]   = ((const float4*)(qg   + tokenBase * 64))[tid];
      ((float4*)sPOS)[tid] = ((const float4*)(posg + tokenBase * 64))[tid];
    }
    sWp[tid & 255] = Wpg[tid & 255];
    if (tid < 64) sBp[tid] = bpg[tid];

    // k tile: 128 rows x 64 f32 -> f16 LDS. 2 threads per row, 16B LDS stores.
    const float4* k4 = (const float4*)(kg + tokenBase * 16 * 64);
    const int r = tid >> 1, half = (tid & 1) * 8;
#pragma unroll
    for (int j = 0; j < 8; j += 2) {
      float4 va = k4[r * 16 + half + j];
      float4 vb = k4[r * 16 + half + j + 1];
      v8h h = { (_Float16)va.x, (_Float16)va.y, (_Float16)va.z, (_Float16)va.w,
                (_Float16)vb.x, (_Float16)vb.y, (_Float16)vb.z, (_Float16)vb.w };
      *(v8h*)(sA + (size_t)(r * 16 + half + j) * 4) = h;
    }
    // speculative prefetch of next tile's k rows (global_prefetch_b8)
    __builtin_prefetch(kg + (tokenBase + TILE_T) * 16 * 64 + (size_t)tid * 64, 0, 0);
  }
  __syncthreads();

  // ---- LayerNorm1(q) -> sQN (f16), rows 0..7 valid ----
  if (tid < TILE_T) {
    float mu = 0.f;
#pragma unroll
    for (int i = 0; i < 64; ++i) mu += sQ[tid * 64 + i];
    mu *= (1.f / 64.f);
    float var = 0.f;
#pragma unroll
    for (int i = 0; i < 64; ++i) { float d = sQ[tid * 64 + i] - mu; var += d * d; }
    float rs = rsqrtf(var * (1.f / 64.f) + EPS);
#pragma unroll
    for (int i = 0; i < 64; ++i)
      sQN[tid * 64 + i] =
          (_Float16)((sQ[tid * 64 + i] - mu) * rs * ln1g[i] + ln1b[i]);
  }
  __syncthreads();

  // ---- qp = LN1(q) @ Wq^T + bq  (waves 0..3, one col-tile each) ----
  if (wid < 4) {
    const int ct = wid;
    const float bc = bq[ct * 16 + l15];
    v8f acc = {};
    acc = wmma16(load_a_frag(sQN, 64, 0, 0, l15, hi),
                 load_b_frag(wWq, 64, ct * 16, 0, l15, hi), acc);
    acc = wmma16(load_a_frag(sQN, 64, 0, 32, l15, hi),
                 load_b_frag(wWq, 64, ct * 16, 32, l15, hi), acc);
    if (hi == 0) {   // only rows 0..7 are real tokens
#pragma unroll
      for (int r = 0; r < 8; ++r)
        sQP[r * 64 + ct * 16 + l15] = acc[r] + bc;
    }
  }
  // no barrier needed before kp: kp reads sA (ready), writes sKP

  // ---- kp = k @ Wk^T + bk  -> sKP (f16, 128x64). One row-tile per wave.
  //      A fragments hoisted: loaded once, reused for all 4 col-tiles. ----
  {
    const int rt = wid;
    const v16h a0 = load_a_frag(sA, 64, rt * 16, 0, l15, hi);
    const v16h a1 = load_a_frag(sA, 64, rt * 16, 32, l15, hi);
#pragma unroll
    for (int ct = 0; ct < 4; ++ct) {
      const float bc = bk[ct * 16 + l15];
      v8f acc = {};
      acc = wmma16(a0, load_b_frag(wWk, 64, ct * 16, 0, l15, hi), acc);
      acc = wmma16(a1, load_b_frag(wWk, 64, ct * 16, 32, l15, hi), acc);
#pragma unroll
      for (int r = 0; r < 8; ++r)
        sKP[(rt * 16 + r + 8 * hi) * 64 + ct * 16 + l15] = (_Float16)(acc[r] + bc);
    }
  }
  __syncthreads();

  // ---- v = kp @ Wv^T + bv  -> sA (f16, overwrites k16) ----
  {
    const int rt = wid;
    const v16h a0 = load_a_frag(sKP, 64, rt * 16, 0, l15, hi);
    const v16h a1 = load_a_frag(sKP, 64, rt * 16, 32, l15, hi);
#pragma unroll
    for (int ct = 0; ct < 4; ++ct) {
      const float bc = bv[ct * 16 + l15];
      v8f acc = {};
      acc = wmma16(a0, load_b_frag(wWv, 64, ct * 16, 0, l15, hi), acc);
      acc = wmma16(a1, load_b_frag(wWv, 64, ct * 16, 32, l15, hi), acc);
#pragma unroll
      for (int r = 0; r < 8; ++r)
        sA[(rt * 16 + r + 8 * hi) * 64 + ct * 16 + l15] = (_Float16)(acc[r] + bc);
    }
  }
  __syncthreads();

  // ---- a = kp - qp + pos@Wp^T + bp ; softmax over M ; o = sum_m a*v ----
  {
    const float4* sPOS4 = (const float4*)sPOS;
    const float4* sWp4  = (const float4*)sWp;
#pragma unroll
    for (int it = 0; it < 2; ++it) {
      const int idx = it * 256 + tid;        // 0..511 = 8 tokens x 64 dims
      const int t = idx >> 6, d = idx & 63;
      const float  qv  = sQP[t * 64 + d];
      const float4 wp  = sWp4[d];
      const float  bpd = sBp[d];
      float av[16], mx = -1e30f;
#pragma unroll
      for (int m = 0; m < 16; ++m) {
        const int r = t * 16 + m;
        const float4 ps = sPOS4[r];
        const float pp = ps.x * wp.x + ps.y * wp.y + ps.z * wp.z + ps.w * wp.w + bpd;
        const float a = (float)sKP[r * 64 + d] - qv + pp;
        av[m] = a; mx = fmaxf(mx, a);
      }
      float s = 0.f;
#pragma unroll
      for (int m = 0; m < 16; ++m) { const float e = __expf(av[m] - mx); av[m] = e; s += e; }
      float o = 0.f;
#pragma unroll
      for (int m = 0; m < 16; ++m) o += av[m] * (float)sA[(t * 16 + m) * 64 + d];
      sO[t * 64 + d] = (_Float16)(o / s);
    }
  }
  __syncthreads();

  // ---- x = q + o @ Wo^T + bo ----
  if (wid < 4) {
    const int ct = wid;
    const float bc = bo[ct * 16 + l15];
    v8f acc = {};
    acc = wmma16(load_a_frag(sO, 64, 0, 0, l15, hi),
                 load_b_frag(wWo, 64, ct * 16, 0, l15, hi), acc);
    acc = wmma16(load_a_frag(sO, 64, 0, 32, l15, hi),
                 load_b_frag(wWo, 64, ct * 16, 32, l15, hi), acc);
    if (hi == 0) {
#pragma unroll
      for (int r = 0; r < 8; ++r) {
        const int c = ct * 16 + l15;
        sX[r * 64 + c] = acc[r] + bc + sQ[r * 64 + c];
      }
    }
  }
  __syncthreads();

  // ---- LayerNorm2(x) -> sXN (f16) ----
  if (tid < TILE_T) {
    float mu = 0.f;
#pragma unroll
    for (int i = 0; i < 64; ++i) mu += sX[tid * 64 + i];
    mu *= (1.f / 64.f);
    float var = 0.f;
#pragma unroll
    for (int i = 0; i < 64; ++i) { float d = sX[tid * 64 + i] - mu; var += d * d; }
    float rs = rsqrtf(var * (1.f / 64.f) + EPS);
#pragma unroll
    for (int i = 0; i < 64; ++i)
      sXN[tid * 64 + i] =
          (_Float16)((sX[tid * 64 + i] - mu) * rs * ln2g[i] + ln2b[i]);
  }
  __syncthreads();

  // ---- FFN1: h1 = relu(xn @ W1^T + b1) -> sKP as [16][256] f16.
  //      A fragments shared by both col-tiles of this wave. ----
  {
    _Float16* sH1 = sKP;
    const v16h a0 = load_a_frag(sXN, 64, 0, 0, l15, hi);
    const v16h a1 = load_a_frag(sXN, 64, 0, 32, l15, hi);
#pragma unroll
    for (int cti = 0; cti < 2; ++cti) {
      const int ct = wid + cti * 8;          // 16 col tiles over F=256
      const float bc = b1[ct * 16 + l15];
      v8f acc = {};
      acc = wmma16(a0, load_b_frag(wW1, 64, ct * 16, 0, l15, hi), acc);
      acc = wmma16(a1, load_b_frag(wW1, 64, ct * 16, 32, l15, hi), acc);
      if (hi == 0) {
#pragma unroll
        for (int r = 0; r < 8; ++r)
          sH1[r * 256 + ct * 16 + l15] = (_Float16)fmaxf(acc[r] + bc, 0.f);
      }
    }
  }
  __syncthreads();

  // ---- FFN2 + residual -> global ----
  if (wid < 4) {
    const _Float16* sH1 = sKP;
    const int ct = wid;
    const float bc = b2[ct * 16 + l15];
    v8f acc = {};
#pragma unroll
    for (int kt = 0; kt < 8; ++kt)           // K = 256
      acc = wmma16(load_a_frag(sH1, 256, 0, kt * 32, l15, hi),
                   load_b_frag(wW2, 256, ct * 16, kt * 32, l15, hi), acc);
    if (hi == 0) {
      const int c = ct * 16 + l15;
#pragma unroll
      for (int r = 0; r < 8; ++r)
        outg[(tokenBase + r) * 64 + c] = acc[r] + bc + sX[r * 64 + c];
    }
  }
}

extern "C" void kernel_launch(void* const* d_in, const int* in_sizes, int n_in,
                              void* d_out, int out_size, void* d_ws, size_t ws_size,
                              hipStream_t stream) {
  (void)in_sizes; (void)n_in; (void)out_size; (void)ws_size;
  const float* q    = (const float*)d_in[0];
  const float* k    = (const float*)d_in[1];
  const float* pos  = (const float*)d_in[2];
  const float* Wq   = (const float*)d_in[3];
  const float* bq   = (const float*)d_in[4];
  const float* Wk   = (const float*)d_in[5];
  const float* bk   = (const float*)d_in[6];
  const float* Wv   = (const float*)d_in[7];
  const float* bv   = (const float*)d_in[8];
  const float* Wp   = (const float*)d_in[9];
  const float* bp   = (const float*)d_in[10];
  const float* Wo   = (const float*)d_in[11];
  const float* bo   = (const float*)d_in[12];
  const float* W1   = (const float*)d_in[13];
  const float* b1   = (const float*)d_in[14];
  const float* W2   = (const float*)d_in[15];
  const float* b2   = (const float*)d_in[16];
  const float* ln1g = (const float*)d_in[17];
  const float* ln1b = (const float*)d_in[18];
  const float* ln2g = (const float*)d_in[19];
  const float* ln2b = (const float*)d_in[20];

  _Float16* wgt = (_Float16*)d_ws;           // 49152 halves = 96 KB
  convert_weights<<<192, 256, 0, stream>>>(Wq, Wk, Wv, Wo, W1, W2, wgt);

  const int nTiles = (8 * 16384) / TILE_T;   // 16384 workgroups
  te_fused<<<nTiles, 256, 0, stream>>>(q, k, pos, bq, bk, bv, Wp, bp, bo, b1, b2,
                                       ln1g, ln1b, ln2g, ln2b, wgt, (float*)d_out);
}
